// GAT0tap_ma_2302102470994
// MI455X (gfx1250) — matrix operand order repaired and checked by reading, not compiled, not run on hardware
//
#include <hip/hip_runtime.h>
#include <hip/hip_bf16.h>

// ---------------------------------------------------------------------------
// GATv2 (2 layers) + MLP head for MI455X (gfx1250), wave32 + WMMA bf16.
//  - node GEMMs: v_wmma_f32_16x16x32_bf16, dual-weight (Wl/Wr share A frags)
//  - edge GEMM fused with GATv2 logit epilogue (ef never materialized)
//  - softmax + aggregation: CSR gather, one wave per (node, head), no big atomics
// ---------------------------------------------------------------------------

#define NNODES 49920
#define NEDGES 200000
#define NGRAPH 128
#define INCH   128
#define EDIM   16
#define EDIMP  32      // padded K for edge GEMM (multiple of 32)
#define HEADS  4
#define H1     1024    // HEADS*C1
#define H2     256     // HEADS*C2
#define C1     256
#define C2     64
#define NEG_SLOPE 0.2f

typedef __attribute__((ext_vector_type(16))) __bf16 v16bf;
typedef __attribute__((ext_vector_type(8)))  __bf16 v8bf;
typedef __attribute__((ext_vector_type(8)))  float  v8f;

// -------- scalar bf16 helpers (storage type = unsigned short, RNE) ----------
__device__ __forceinline__ unsigned short f2bf(float f) {
  unsigned u = __float_as_uint(f);
  unsigned r = (u + 0x7FFFu + ((u >> 16) & 1u)) >> 16;
  return (unsigned short)r;
}
__device__ __forceinline__ float bf2f(unsigned short s) {
  return __uint_as_float(((unsigned)s) << 16);
}

// ---------------------------------------------------------------------------
// fp32 rows -> bf16 rows with optional zero-padded columns
// ---------------------------------------------------------------------------
__global__ void cvt_rows_k(const float* __restrict__ src, unsigned short* __restrict__ dst,
                           long long rows, int cols, int dcols) {
  long long i = (long long)blockIdx.x * blockDim.x + threadIdx.x;
  long long total = rows * (long long)dcols;
  if (i >= total) return;
  long long r = i / dcols;
  int c = (int)(i - r * dcols);
  float v = (c < cols) ? src[r * cols + c] : 0.0f;
  dst[i] = f2bf(v);
}

// ---------------------------------------------------------------------------
// Pack fp32 W[K,N] into WMMA B-fragment layout:
//   Wp[((kt*2+half)*N + n)*8 + v] = {bf16 W[k0,n] (lo), bf16 W[k0+1,n] (hi)},
//   k0 = kt*32 + half*16 + 2v.  Lane (n=lane&15, half=lane>>4) reads its full
//   16-element fragment as one contiguous 32B load.
// ---------------------------------------------------------------------------
__global__ void pack_w_k(const float* __restrict__ W, unsigned* __restrict__ Wp,
                         int K, int Kp, int Nn) {
  int i = blockIdx.x * blockDim.x + threadIdx.x;
  int total = (Kp >> 5) * 2 * Nn * 8;
  if (i >= total) return;
  int v  = i & 7;
  int t  = i >> 3;
  int n  = t % Nn;
  int th = t / Nn;              // kt*2 + half
  int hf = th & 1;
  int kt = th >> 1;
  int k0 = kt * 32 + hf * 16 + 2 * v;
  float w0 = (k0     < K) ? W[(size_t)k0       * Nn + n] : 0.0f;
  float w1 = (k0 + 1 < K) ? W[(size_t)(k0 + 1) * Nn + n] : 0.0f;
  Wp[i] = ((unsigned)f2bf(w1) << 16) | (unsigned)f2bf(w0);
}

// ---------------------------------------------------------------------------
// Dual node GEMM: Cl/Cr[M,N](bf16) = A[M,Kp](bf16) @ {Wl,Wr} + bias.
// One wave = one 16x16 tile; 4-wave block covers 16x64; A frags shared.
// Requires M%16==0, Nn%64==0, Kp%32==0.
// ---------------------------------------------------------------------------
__global__ void gemm_dual_k(const unsigned short* __restrict__ A, int Kp,
                            const unsigned* __restrict__ Wp1, const float* __restrict__ bias1,
                            unsigned short* __restrict__ Cout1,
                            const unsigned* __restrict__ Wp2, const float* __restrict__ bias2,
                            unsigned short* __restrict__ Cout2,
                            int Nn) {
  const int lane = threadIdx.x & 31;
  const int wave = threadIdx.x >> 5;
  const long long m0 = (long long)blockIdx.x * 16;
  const int n0 = blockIdx.y * 64 + wave * 16;
  const int mr = lane & 15;
  const int hf = lane >> 4;

  const unsigned short* arow = A + (m0 + mr) * (long long)Kp + hf * 16;

  v8f acc1 = {};
  v8f acc2 = {};
  const int ktiles = Kp >> 5;
  for (int kt = 0; kt < ktiles; ++kt) {
    v8bf alo = *(const v8bf*)(arow + kt * 32);
    v8bf ahi = *(const v8bf*)(arow + kt * 32 + 8);
    v16bf a = __builtin_shufflevector(alo, ahi,
                                      0, 1, 2, 3, 4, 5, 6, 7,
                                      8, 9, 10, 11, 12, 13, 14, 15);
    const unsigned* bp1 = Wp1 + ((size_t)(kt * 2 + hf) * Nn + (n0 + mr)) * 8;
    v16bf b1 = *(const v16bf*)bp1;
    acc1 = __builtin_amdgcn_wmma_f32_16x16x32_bf16(false, a, false, b1,
                                                   (short)0, acc1, false, false);
    const unsigned* bp2 = Wp2 + ((size_t)(kt * 2 + hf) * Nn + (n0 + mr)) * 8;
    v16bf b2 = *(const v16bf*)bp2;
    acc2 = __builtin_amdgcn_wmma_f32_16x16x32_bf16(false, a, false, b2,
                                                   (short)0, acc2, false, false);
  }

  const int n = n0 + mr;
  const long long mbase = m0 + hf * 8;
  const float bs1 = bias1[n];
  const float bs2 = bias2[n];
#pragma unroll
  for (int r = 0; r < 8; ++r) {
    Cout1[(mbase + r) * Nn + n] = f2bf(acc1[r] + bs1);
    Cout2[(mbase + r) * Nn + n] = f2bf(acc2[r] + bs2);
  }
}

// ---------------------------------------------------------------------------
// Fused edge GEMM + GATv2 logit epilogue.
//   ef_tile = edge_attr_tile @ We      (one WMMA, Kp = 32)
//   v       = leaky_relu(ef + xl[src] + xr[dst])
//   lg[e,h] += sum_n att[n] * v[e,n]   (16-lane shfl reduction + atomicAdd)
// lg must be pre-zeroed. C/D layout: element r -> edge m0+hf*8+r, chan n0+mr.
// ---------------------------------------------------------------------------
__global__ void edge_gemm_logits_k(const unsigned short* __restrict__ eb,   // [E, 32] bf16
                                   const unsigned* __restrict__ Wep,        // packed [32, Nn]
                                   const unsigned short* __restrict__ xl,   // [N, Nn] bf16
                                   const unsigned short* __restrict__ xr,   // [N, Nn] bf16
                                   const int* __restrict__ src, const int* __restrict__ dst,
                                   const float* __restrict__ att,           // [Nn] flat [H][C]
                                   float* __restrict__ lg,                  // [E*HEADS]
                                   int Nn, int C) {
  const int lane = threadIdx.x & 31;
  const int wave = threadIdx.x >> 5;
  const long long m0 = (long long)blockIdx.x * 16;
  const int n0 = blockIdx.y * 64 + wave * 16;
  const int mr = lane & 15;
  const int hf = lane >> 4;

  const unsigned short* arow = eb + (m0 + mr) * EDIMP + hf * 16;
  v8bf alo = *(const v8bf*)(arow);
  v8bf ahi = *(const v8bf*)(arow + 8);
  v16bf a = __builtin_shufflevector(alo, ahi,
                                    0, 1, 2, 3, 4, 5, 6, 7,
                                    8, 9, 10, 11, 12, 13, 14, 15);
  const unsigned* bp = Wep + ((size_t)hf * Nn + (n0 + mr)) * 8;
  v16bf b = *(const v16bf*)bp;
  v8f acc = {};
  acc = __builtin_amdgcn_wmma_f32_16x16x32_bf16(false, a, false, b,
                                                (short)0, acc, false, false);

  const int n = n0 + mr;          // global channel; 16-chunk stays in one head
  const int h = n / C;
  const float an = att[n];
#pragma unroll
  for (int r = 0; r < 8; ++r) {
    long long e = m0 + hf * 8 + r;
    int s = src[e], d = dst[e];
    float v = acc[r] + bf2f(xl[(size_t)s * Nn + n]) + bf2f(xr[(size_t)d * Nn + n]);
    v = (v > 0.0f) ? v : NEG_SLOPE * v;
    float c = an * v;
#pragma unroll
    for (int off = 8; off > 0; off >>= 1) c += __shfl_xor(c, off, 16);
    if (mr == 0) atomicAdd(lg + e * HEADS + h, c);
  }
}

// ---------------------------------------------------------------------------
// zero fill
// ---------------------------------------------------------------------------
__global__ void zero_u32_k(unsigned* __restrict__ p, long long n) {
  long long i = (long long)blockIdx.x * blockDim.x + threadIdx.x;
  if (i < n) p[i] = 0u;
}

// ---------------------------------------------------------------------------
// CSR build: in-degree count, single-block LDS scan, fill.
// ---------------------------------------------------------------------------
__global__ void deg_count_k(const int* __restrict__ dst, int* __restrict__ deg, int E) {
  int i = blockIdx.x * blockDim.x + threadIdx.x;
  if (i < E) atomicAdd(deg + dst[i], 1);
}

__global__ void scan_k(const int* __restrict__ deg, int* __restrict__ rowptr, int n) {
  __shared__ int buf[1024];
  __shared__ int carry;
  if (threadIdx.x == 0) carry = 0;
  __syncthreads();
  for (int base = 0; base < n; base += 1024) {
    int i = base + threadIdx.x;
    int v = (i < n) ? deg[i] : 0;
    buf[threadIdx.x] = v;
    __syncthreads();
#pragma unroll
    for (int off = 1; off < 1024; off <<= 1) {
      int t = (threadIdx.x >= (unsigned)off) ? buf[threadIdx.x - off] : 0;
      __syncthreads();
      buf[threadIdx.x] += t;
      __syncthreads();
    }
    if (i < n) rowptr[i] = carry + buf[threadIdx.x] - v;   // exclusive
    __syncthreads();
    if (threadIdx.x == 1023) carry += buf[1023];
    __syncthreads();
  }
  if (threadIdx.x == 0) rowptr[n] = carry;
}

__global__ void csr_fill_k(const int* __restrict__ dst, const int* __restrict__ rowptr,
                           int* __restrict__ cursor, int* __restrict__ eidx, int E) {
  int i = blockIdx.x * blockDim.x + threadIdx.x;
  if (i >= E) return;
  int d = dst[i];
  int p = atomicAdd(cursor + d, 1);
  eidx[rowptr[d] + p] = i;
}

// ---------------------------------------------------------------------------
// Fused segment softmax + aggregation + bias + relu, gather form.
// One wave per (node, head): max over incoming logits, then unnormalized
// exp-weighted channel accumulation + denominator, final 1/(sum+eps) scale.
// ---------------------------------------------------------------------------
template <int C, bool BF16OUT>
__global__ void agg_gather_k(const int* __restrict__ rowptr, const int* __restrict__ eidx,
                             const int* __restrict__ src, const float* __restrict__ lg,
                             const unsigned short* __restrict__ xl,   // [N, HEADS*C]
                             const float* __restrict__ bias,          // [HEADS*C]
                             unsigned short* __restrict__ out_bf,
                             float* __restrict__ out_f) {
  int gw = blockIdx.x * (blockDim.x >> 5) + (threadIdx.x >> 5);   // node*HEADS + h
  if (gw >= NNODES * HEADS) return;
  int node = gw >> 2;
  int h = gw & 3;
  int lane = threadIdx.x & 31;
  int beg = rowptr[node], end = rowptr[node + 1];

  float mx = -3.402823e38f;
  for (int j = beg; j < end; ++j)
    mx = fmaxf(mx, lg[(size_t)eidx[j] * HEADS + h]);

  constexpr int NC = C / 32;
  float acc[NC];
#pragma unroll
  for (int q = 0; q < NC; ++q) acc[q] = 0.0f;
  float dsum = 0.0f;

  for (int j = beg; j < end; ++j) {
    int e = eidx[j];
    float w = __expf(lg[(size_t)e * HEADS + h] - mx);
    dsum += w;
    const unsigned short* px = xl + (size_t)src[e] * (HEADS * C) + h * C;
#pragma unroll
    for (int q = 0; q < NC; ++q) acc[q] += w * bf2f(px[lane + q * 32]);
  }

  float inv = 1.0f / (dsum + 1e-16f);
  size_t ob = (size_t)node * (HEADS * C) + h * C;
#pragma unroll
  for (int q = 0; q < NC; ++q) {
    float v = acc[q] * inv + bias[h * C + lane + q * 32];
    v = (v > 0.0f) ? v : 0.0f;
    if (BF16OUT) out_bf[ob + lane + q * 32] = f2bf(v);
    else         out_f [ob + lane + q * 32] = v;
  }
}

// ---------------------------------------------------------------------------
// MLP head on master nodes (last node of each graph): 128 blocks x 64 thr.
// ---------------------------------------------------------------------------
__global__ void head_k(const float* __restrict__ h2, const int* __restrict__ n_nodes,
                       const float* __restrict__ fc1w, const float* __restrict__ fc1b,
                       const float* __restrict__ fc2w, const float* __restrict__ fc2b,
                       float* __restrict__ out) {
  __shared__ float z[64];
  __shared__ long long row;
  int b = blockIdx.x;
  int t = threadIdx.x;
  if (t == 0) {
    long long cum = 0;
    for (int i = 0; i <= b; ++i) cum += (long long)n_nodes[i];
    row = cum - 1;
  }
  __syncthreads();
  const float* m = h2 + row * H2;
  float s = fc1b[t];
  for (int k = 0; k < H2; ++k) s += m[k] * fc1w[k * 64 + t];
  z[t] = (s > 0.0f) ? s : 0.0f;
  __syncthreads();
  if (t == 0) {
    float o = fc2b[0];
    for (int k = 0; k < 64; ++k) o += z[k] * fc2w[k];
    out[b] = o;
  }
}

// ---------------------------------------------------------------------------
// launcher
// ---------------------------------------------------------------------------
extern "C" void kernel_launch(void* const* d_in, const int* in_sizes, int n_in,
                              void* d_out, int out_size, void* d_ws, size_t ws_size,
                              hipStream_t stream) {
  (void)in_sizes; (void)n_in; (void)out_size; (void)ws_size;

  const float* x    = (const float*)d_in[0];
  const int*   ei   = (const int*)d_in[1];
  const float* ea   = (const float*)d_in[2];
  const int*   nnod = (const int*)d_in[3];
  const float* Wl1  = (const float*)d_in[4];
  const float* bl1  = (const float*)d_in[5];
  const float* Wr1  = (const float*)d_in[6];
  const float* br1  = (const float*)d_in[7];
  const float* We1  = (const float*)d_in[8];
  const float* att1 = (const float*)d_in[9];
  const float* b1   = (const float*)d_in[10];
  const float* Wl2  = (const float*)d_in[11];
  const float* bl2  = (const float*)d_in[12];
  const float* Wr2  = (const float*)d_in[13];
  const float* br2  = (const float*)d_in[14];
  const float* We2  = (const float*)d_in[15];
  const float* att2 = (const float*)d_in[16];
  const float* b2   = (const float*)d_in[17];
  const float* fc1w = (const float*)d_in[18];
  const float* fc1b = (const float*)d_in[19];
  const float* fc2w = (const float*)d_in[20];
  const float* fc2b = (const float*)d_in[21];
  float* out = (float*)d_out;

  const int* src = ei;            // edge_index row 0
  const int* dst = ei + NEDGES;   // edge_index row 1

  // ---- workspace carve-out (~0.45 GB) ----
  char* ws = (char*)d_ws;
  size_t off = 0;
  auto take = [&](size_t bytes) -> char* {
    char* p = ws + off;
    off = (off + bytes + 255) & ~(size_t)255;
    return p;
  };
  unsigned short* xb     = (unsigned short*)take((size_t)NNODES * INCH * 2);
  unsigned short* eb     = (unsigned short*)take((size_t)NEDGES * EDIMP * 2);
  unsigned*       Wl1p   = (unsigned*)take((size_t)INCH  * H1 * 2);
  unsigned*       Wr1p   = (unsigned*)take((size_t)INCH  * H1 * 2);
  unsigned*       We1p   = (unsigned*)take((size_t)EDIMP * H1 * 2);
  unsigned*       Wl2p   = (unsigned*)take((size_t)H1    * H2 * 2);
  unsigned*       Wr2p   = (unsigned*)take((size_t)H1    * H2 * 2);
  unsigned*       We2p   = (unsigned*)take((size_t)EDIMP * H2 * 2);
  unsigned short* xl1    = (unsigned short*)take((size_t)NNODES * H1 * 2);
  unsigned short* xr1    = (unsigned short*)take((size_t)NNODES * H1 * 2);
  unsigned short* h1b    = (unsigned short*)take((size_t)NNODES * H1 * 2);
  unsigned short* xl2    = (unsigned short*)take((size_t)NNODES * H2 * 2);
  unsigned short* xr2    = (unsigned short*)take((size_t)NNODES * H2 * 2);
  float*          h2     = (float*)take((size_t)NNODES * H2 * 4);
  float*          lg     = (float*)take((size_t)NEDGES * HEADS * 4);
  int*            deg    = (int*)take((size_t)NNODES * 4);
  int*            rowptr = (int*)take((size_t)(NNODES + 1) * 4);
  int*            cursor = (int*)take((size_t)NNODES * 4);
  int*            eidx   = (int*)take((size_t)NEDGES * 4);

  const int TPB = 256;
  auto blocks = [](long long total, int tpb) { return (unsigned)((total + tpb - 1) / tpb); };

  // ---- stage 0: conversions & weight packing ----
  cvt_rows_k<<<blocks((long long)NNODES * INCH, TPB), TPB, 0, stream>>>(x, xb, NNODES, INCH, INCH);
  cvt_rows_k<<<blocks((long long)NEDGES * EDIMP, TPB), TPB, 0, stream>>>(ea, eb, NEDGES, EDIM, EDIMP);
  pack_w_k<<<blocks((long long)INCH  * H1 / 2, TPB), TPB, 0, stream>>>(Wl1, Wl1p, INCH, INCH, H1);
  pack_w_k<<<blocks((long long)INCH  * H1 / 2, TPB), TPB, 0, stream>>>(Wr1, Wr1p, INCH, INCH, H1);
  pack_w_k<<<blocks((long long)EDIMP * H1 / 2, TPB), TPB, 0, stream>>>(We1, We1p, EDIM, EDIMP, H1);
  pack_w_k<<<blocks((long long)H1    * H2 / 2, TPB), TPB, 0, stream>>>(Wl2, Wl2p, H1, H1, H2);
  pack_w_k<<<blocks((long long)H1    * H2 / 2, TPB), TPB, 0, stream>>>(Wr2, Wr2p, H1, H1, H2);
  pack_w_k<<<blocks((long long)EDIMP * H2 / 2, TPB), TPB, 0, stream>>>(We2, We2p, EDIM, EDIMP, H2);

  // ---- CSR of incoming edges (shared by both layers) ----
  zero_u32_k<<<blocks(NNODES, TPB), TPB, 0, stream>>>((unsigned*)deg, NNODES);
  zero_u32_k<<<blocks(NNODES, TPB), TPB, 0, stream>>>((unsigned*)cursor, NNODES);
  deg_count_k<<<blocks(NEDGES, TPB), TPB, 0, stream>>>(dst, deg, NEDGES);
  scan_k<<<1, 1024, 0, stream>>>(deg, rowptr, NNODES);
  csr_fill_k<<<blocks(NEDGES, TPB), TPB, 0, stream>>>(dst, rowptr, cursor, eidx, NEDGES);

  const unsigned aggBlocks = blocks((long long)NNODES * HEADS * 32, TPB);

  // ---- layer 1 ----
  gemm_dual_k<<<dim3(NNODES / 16, H1 / 64), 128, 0, stream>>>(
      xb, INCH, Wl1p, bl1, xl1, Wr1p, br1, xr1, H1);
  zero_u32_k<<<blocks((long long)NEDGES * HEADS, TPB), TPB, 0, stream>>>((unsigned*)lg, (long long)NEDGES * HEADS);
  edge_gemm_logits_k<<<dim3(NEDGES / 16, H1 / 64), 128, 0, stream>>>(
      eb, We1p, xl1, xr1, src, dst, att1, lg, H1, C1);
  agg_gather_k<C1, true><<<aggBlocks, TPB, 0, stream>>>(
      rowptr, eidx, src, lg, xl1, b1, h1b, nullptr);

  // ---- layer 2 ----
  gemm_dual_k<<<dim3(NNODES / 16, H2 / 64), 128, 0, stream>>>(
      h1b, H1, Wl2p, bl2, xl2, Wr2p, br2, xr2, H2);
  zero_u32_k<<<blocks((long long)NEDGES * HEADS, TPB), TPB, 0, stream>>>((unsigned*)lg, (long long)NEDGES * HEADS);
  edge_gemm_logits_k<<<dim3(NEDGES / 16, H2 / 64), 128, 0, stream>>>(
      eb, We2p, xl2, xr2, src, dst, att2, lg, H2, C2);
  agg_gather_k<C2, false><<<aggBlocks, TPB, 0, stream>>>(
      rowptr, eidx, src, lg, xl2, b2, nullptr, h2);

  // ---- MLP head ----
  head_k<<<NGRAPH, 64, 0, stream>>>(h2, nnod, fc1w, fc1b, fc2w, fc2b, out);
}
